// DynamicHyperGraphAttention_16106127360034
// MI455X (gfx1250) — matrix operand order repaired and checked by reading, not compiled, not run on hardware
//
#include <hip/hip_runtime.h>
#include <hip/hip_bf16.h>

typedef __attribute__((ext_vector_type(16))) _Float16 v16h;
typedef __attribute__((ext_vector_type(8)))  float    v8f;

#define HID_   128
#define NHEADS_ 4
#define DHEAD_  32
#define KNN_    10

// ---------------------------------------------------------------------------
// WMMA helpers (gfx1250, wave32). D = A(16x32 f16) * B(32x16 f16) + C(16x16 f32)
// ---------------------------------------------------------------------------
__device__ __forceinline__ v8f wmma_f16(v16h a, v16h b, v8f c) {
  return __builtin_amdgcn_wmma_f32_16x16x32_f16(false, a, false, b, (short)0, c,
                                                false, false);
}

// A-matrix fragment: row-major f16 tile, origin at (row0, k0), leading dim lda.
// Lane l: h = l>>4, r = l&15 holds row r; element e maps to K = h*16 + e.
__device__ __forceinline__ v16h load_a_frag(const _Float16* A, int lda) {
  int lane = threadIdx.x & 31;
  int h = lane >> 4, r = lane & 15;
  v16h a;
#pragma unroll
  for (int e = 0; e < 16; ++e) {
    int k = (h << 4) + e;
    a[e] = A[(size_t)r * lda + k];
  }
  return a;
}

// B fragment where B[k][n] = W[n][k] (W row-major [N x K], tile origin (n0,k0)).
// Lane l: h = l>>4, c = l&15 holds column c; element e maps to K = h*16 + e.
__device__ __forceinline__ v16h load_b_frag_T(const _Float16* W, int ldw) {
  int lane = threadIdx.x & 31;
  int h = lane >> 4, c = lane & 15;
  v16h b;
#pragma unroll
  for (int e = 0; e < 16; ++e)
    b[e] = W[(size_t)c * ldw + (h << 4) + e];
  return b;
}

// B fragment, B row-major [K x N] tile (origin (k0,n0)).
__device__ __forceinline__ v16h load_b_frag(const _Float16* B, int ldb) {
  int lane = threadIdx.x & 31;
  int h = lane >> 4, c = lane & 15;
  v16h b;
#pragma unroll
  for (int e = 0; e < 16; ++e)
    b[e] = B[(size_t)((h << 4) + e) * ldb + c];
  return b;
}

// ---------------------------------------------------------------------------
// Specialized 128x128 GEMM: out[M,128] = A[M,128] @ W[128,128]^T + bias.
// One wave computes a 16-row x 64-col strip: A-fragments loaded once, B-fragments
// software-pipelined one step ahead of the consuming WMMA. grid = (M/16, 2).
// ---------------------------------------------------------------------------
template <bool F16OUT>
__global__ void gemm128_kernel(const _Float16* __restrict__ A,
                               const _Float16* __restrict__ W,
                               const float* __restrict__ bias,
                               void* __restrict__ out) {
  int m0 = blockIdx.x << 4;
  int nbase = blockIdx.y << 6;  // 4 column tiles of 16
  int lane = threadIdx.x & 31;
  int h = lane >> 4, c = lane & 15;

  v16h a[4];
#pragma unroll
  for (int k = 0; k < 4; ++k)
    a[k] = load_a_frag(A + (size_t)m0 * HID_ + 32 * k, HID_);

  v16h bn = load_b_frag_T(W + (size_t)nbase * HID_, HID_);
#pragma unroll
  for (int n = 0; n < 4; ++n) {
    v8f acc = {};
#pragma unroll
    for (int k = 0; k < 4; ++k) {
      v16h bc = bn;
      if (!(n == 3 && k == 3)) {
        int nk = n * 4 + k + 1;
        int nn_ = nk >> 2, kk_ = nk & 3;
        bn = load_b_frag_T(W + (size_t)(nbase + nn_ * 16) * HID_ + 32 * kk_, HID_);
      }
      acc = wmma_f16(a[k], bc, acc);
    }
    int col = nbase + n * 16 + c;
    float bv = bias ? bias[col] : 0.0f;
#pragma unroll
    for (int v = 0; v < 8; ++v) {
      size_t row = (size_t)(m0 + v + (h << 3));
      float val = acc[v] + bv;
      if (F16OUT)
        ((_Float16*)out)[row * HID_ + col] = (_Float16)val;
      else
        ((float*)out)[row * HID_ + col] = val;
    }
  }
}

// ---------------------------------------------------------------------------
// kNN: per 16-row block, stream gram tiles via WMMA with double-buffered
// B-fragments, keep top-10 smallest d2 per row.
// ---------------------------------------------------------------------------
__global__ void knn_topk_kernel(const _Float16* __restrict__ xh,
                                const float* __restrict__ sq,
                                int* __restrict__ nn_idx, int N) {
  __shared__ float tile[16][17];
  __shared__ float cd[16][20];
  __shared__ int   ci[16][20];
  int r0 = blockIdx.x << 4;
  int lane = threadIdx.x;
  int h = lane >> 4, c = lane & 15;
  int myrow = lane >> 1;
  int cbase = (lane & 1) << 3;
  int slot = (lane & 1) * KNN_;

  v16h a0 = load_a_frag(xh + (size_t)r0 * HID_ + 0, HID_);
  v16h a1 = load_a_frag(xh + (size_t)r0 * HID_ + 32, HID_);
  v16h a2 = load_a_frag(xh + (size_t)r0 * HID_ + 64, HID_);
  v16h a3 = load_a_frag(xh + (size_t)r0 * HID_ + 96, HID_);

  float sqr[8];
#pragma unroll
  for (int v = 0; v < 8; ++v) sqr[v] = sq[r0 + v + (h << 3)];

  float best[KNN_]; int bidx[KNN_];
#pragma unroll
  for (int t = 0; t < KNN_; ++t) { best[t] = 3.0e38f; bidx[t] = 0; }

  // prime B fragments for tile j0 = 0
  v16h b0 = load_b_frag_T(xh + 0, HID_);
  v16h b1 = load_b_frag_T(xh + 32, HID_);
  v16h b2 = load_b_frag_T(xh + 64, HID_);
  v16h b3 = load_b_frag_T(xh + 96, HID_);

  for (int j0 = 0; j0 < N; j0 += 16) {
    v8f acc = {};
    acc = wmma_f16(a0, b0, acc);
    acc = wmma_f16(a1, b1, acc);
    acc = wmma_f16(a2, b2, acc);
    acc = wmma_f16(a3, b3, acc);

    // issue next tile's fragment loads; they overlap the WMMAs and the top-k
    // maintenance below
    if (j0 + 16 < N) {
      const _Float16* nb = xh + (size_t)(j0 + 16) * HID_;
      b0 = load_b_frag_T(nb + 0, HID_);
      b1 = load_b_frag_T(nb + 32, HID_);
      b2 = load_b_frag_T(nb + 64, HID_);
      b3 = load_b_frag_T(nb + 96, HID_);
    }

    float sqc = sq[j0 + c];
    __syncthreads();
#pragma unroll
    for (int v = 0; v < 8; ++v)
      tile[v + (h << 3)][c] = sqr[v] + sqc - 2.0f * acc[v];
    __syncthreads();
#pragma unroll
    for (int t = 0; t < 8; ++t) {
      float d = tile[myrow][cbase + t];
      if (d < best[KNN_ - 1]) {
        int idx = j0 + cbase + t;
        int p = KNN_ - 1;
        while (p > 0 && best[p - 1] > d) {
          best[p] = best[p - 1]; bidx[p] = bidx[p - 1]; --p;
        }
        best[p] = d; bidx[p] = idx;
      }
    }
  }
  __syncthreads();
#pragma unroll
  for (int t = 0; t < KNN_; ++t) { cd[myrow][slot + t] = best[t]; ci[myrow][slot + t] = bidx[t]; }
  __syncthreads();
  if (lane < 16) {
    int i = 0, j = 0;
    for (int t = 0; t < KNN_; ++t) {
      if (cd[lane][i] <= cd[lane][KNN_ + j]) { nn_idx[(size_t)(r0 + lane) * KNN_ + t] = ci[lane][i]; ++i; }
      else                                   { nn_idx[(size_t)(r0 + lane) * KNN_ + t] = ci[lane][KNN_ + j]; ++j; }
    }
  }
}

// ---------------------------------------------------------------------------
// Flash attention: one wave per (16-query tile, head). Online softmax; K and V
// fragments issued ahead of their consumers to overlap softmax + LDS staging.
// ---------------------------------------------------------------------------
__global__ void flash_attn_kernel(const _Float16* __restrict__ Q,
                                  const _Float16* __restrict__ Kh,
                                  const _Float16* __restrict__ Vh,
                                  _Float16* __restrict__ O, int Lk, float scale) {
  __shared__ _Float16 pbuf[16][32];
  int q0 = blockIdx.x << 4;
  int hd = blockIdx.y;
  int lane = threadIdx.x;
  int h = lane >> 4, c = lane & 15;
  const int off = hd * DHEAD_;

  v16h qf = load_a_frag(Q + (size_t)q0 * HID_ + off, HID_);

  v8f o0 = {}, o1 = {};
  float rm[8], rl[8];
#pragma unroll
  for (int v = 0; v < 8; ++v) { rm[v] = -3.0e38f; rl[v] = 0.0f; }

  // prime K fragments for first 32-key step
  v16h kb0 = load_b_frag_T(Kh + off, HID_);
  v16h kb1 = load_b_frag_T(Kh + (size_t)16 * HID_ + off, HID_);

  for (int k0 = 0; k0 < Lk; k0 += 32) {
    const _Float16* vb = Vh + (size_t)k0 * HID_ + off;

    v8f s0 = {}, s1 = {};
    s0 = wmma_f16(qf, kb0, s0);
    s1 = wmma_f16(qf, kb1, s1);

    // issue V-fragment loads and next step's K-fragment loads now; they
    // overlap the softmax math and the P round-trip through LDS below
    v16h vf0 = load_b_frag(vb, HID_);
    v16h vf1 = load_b_frag(vb + 16, HID_);
    if (k0 + 32 < Lk) {
      const _Float16* nk = Kh + (size_t)(k0 + 32) * HID_ + off;
      kb0 = load_b_frag_T(nk, HID_);
      kb1 = load_b_frag_T(nk + (size_t)16 * HID_, HID_);
    }

    float f[8];
#pragma unroll
    for (int v = 0; v < 8; ++v) {
      float x0 = s0[v] * scale, x1 = s1[v] * scale;
      float m = fmaxf(x0, x1);
      m = fmaxf(m, __shfl_xor(m, 1, 32));
      m = fmaxf(m, __shfl_xor(m, 2, 32));
      m = fmaxf(m, __shfl_xor(m, 4, 32));
      m = fmaxf(m, __shfl_xor(m, 8, 32));
      float nm = fmaxf(rm[v], m);
      float fv = __expf(rm[v] - nm);
      float p0 = __expf(x0 - nm), p1 = __expf(x1 - nm);
      float ps = p0 + p1;
      ps += __shfl_xor(ps, 1, 32);
      ps += __shfl_xor(ps, 2, 32);
      ps += __shfl_xor(ps, 4, 32);
      ps += __shfl_xor(ps, 8, 32);
      rl[v] = rl[v] * fv + ps;
      rm[v] = nm;
      f[v] = fv;
      s0[v] = p0; s1[v] = p1;
    }
    __syncthreads();
#pragma unroll
    for (int v = 0; v < 8; ++v) {
      pbuf[v + (h << 3)][c]      = (_Float16)s0[v];
      pbuf[v + (h << 3)][16 + c] = (_Float16)s1[v];
    }
    __syncthreads();
    v16h pf = load_a_frag(&pbuf[0][0], 32);
#pragma unroll
    for (int v = 0; v < 8; ++v) { o0[v] *= f[v]; o1[v] *= f[v]; }
    o0 = wmma_f16(pf, vf0, o0);
    o1 = wmma_f16(pf, vf1, o1);
  }
#pragma unroll
  for (int v = 0; v < 8; ++v) {
    float inv = 1.0f / rl[v];
    size_t row = (size_t)(q0 + v + (h << 3));
    O[row * HID_ + off + c]      = (_Float16)(o0[v] * inv);
    O[row * HID_ + off + 16 + c] = (_Float16)(o1[v] * inv);
  }
}

// ---------------------------------------------------------------------------
// Elementwise / reduction / scatter helpers
// ---------------------------------------------------------------------------
__global__ void concat_kernel(const float* a, const float* b, float* x, int N1) {
  int row = blockIdx.x, t = threadIdx.x;
  float v = (row < N1) ? a[(size_t)row * HID_ + t] : b[(size_t)(row - N1) * HID_ + t];
  x[(size_t)row * HID_ + t] = v;
}

__global__ void cvt_f16_kernel(const float* src, _Float16* dst, int n) {
  int i = blockIdx.x * blockDim.x + threadIdx.x;
  if (i < n) dst[i] = (_Float16)src[i];
}

__global__ void row_sq_kernel(const float* x, float* sq) {
  __shared__ float red[HID_];
  int row = blockIdx.x, t = threadIdx.x;
  float v = x[(size_t)row * HID_ + t];
  red[t] = v * v;
  __syncthreads();
  for (int s = HID_ / 2; s > 0; s >>= 1) {
    if (t < s) red[t] += red[t + s];
    __syncthreads();
  }
  if (t == 0) sq[row] = red[0];
}

__global__ void add_knn_kernel(float* x, const float* hf, const int* nn) {
  int row = blockIdx.x, t = threadIdx.x;
  float s = 0.f;
#pragma unroll
  for (int k = 0; k < KNN_; ++k)
    s += hf[(size_t)nn[(size_t)row * KNN_ + k] * HID_ + t];
  x[(size_t)row * HID_ + t] += s;
}

__global__ void zero_kernel(float* p, int n) {
  int i = blockIdx.x * blockDim.x + threadIdx.x;
  if (i < n) p[i] = 0.f;
}

__global__ void deg_kernel(const int* edge, float* D, float* B, int E) {
  int e = blockIdx.x * blockDim.x + threadIdx.x;
  if (e < E) {
    atomicAdd(&D[edge[e]], 1.0f);
    atomicAdd(&B[edge[E + e]], 1.0f);
  }
}

__global__ void scatter_n2e_kernel(const float* xw, const int* edge, int E, float* ef) {
  int e = blockIdx.x, t = threadIdx.x;
  int node = edge[e], he = edge[E + e];
  atomicAdd(&ef[(size_t)he * HID_ + t], xw[(size_t)node * HID_ + t]);
}

__global__ void scale_rows_kernel(float* m, const float* deg) {
  int row = blockIdx.x, t = threadIdx.x;
  float d = deg[row];
  float inv = d > 0.f ? 1.0f / d : 0.f;
  m[(size_t)row * HID_ + t] *= inv;
}

__global__ void scatter_e2n_kernel(const float* ef, const int* edge, int E, float* out) {
  int e = blockIdx.x, t = threadIdx.x;
  int node = edge[e], he = edge[E + e];
  atomicAdd(&out[(size_t)node * HID_ + t], ef[(size_t)he * HID_ + t]);
}

__global__ void scale_bias_kernel(float* m, const float* deg, const float* bias) {
  int row = blockIdx.x, t = threadIdx.x;
  float d = deg[row];
  float inv = d > 0.f ? 1.0f / d : 0.f;
  m[(size_t)row * HID_ + t] = m[(size_t)row * HID_ + t] * inv + bias[t];
}

__global__ void bn_stats_kernel(const float* x, int N, float* mean, float* var) {
  __shared__ float s1[256], s2[256];
  int col = blockIdx.x, t = threadIdx.x;
  float a = 0.f, b = 0.f;
  for (int i = t; i < N; i += 256) {
    float v = x[(size_t)i * HID_ + col];
    a += v; b += v * v;
  }
  s1[t] = a; s2[t] = b;
  __syncthreads();
  for (int s = 128; s > 0; s >>= 1) {
    if (t < s) { s1[t] += s1[t + s]; s2[t] += s2[t + s]; }
    __syncthreads();
  }
  if (t == 0) {
    float m = s1[0] / N;
    mean[col] = m;
    var[col] = s2[0] / N - m * m;
  }
}

__global__ void bn_elu_kernel(const float* x, _Float16* yh, const float* mean,
                              const float* var, const float* gamma, const float* beta) {
  int row = blockIdx.x, t = threadIdx.x;
  float v = (x[(size_t)row * HID_ + t] - mean[t]) * rsqrtf(var[t] + 1e-5f) * gamma[t] + beta[t];
  v = v > 0.f ? v : (__expf(v) - 1.0f);
  yh[(size_t)row * HID_ + t] = (_Float16)v;
}

// ---------------------------------------------------------------------------
// Orchestration
// ---------------------------------------------------------------------------
extern "C" void kernel_launch(void* const* d_in, const int* in_sizes, int n_in,
                              void* d_out, int out_size, void* d_ws, size_t ws_size,
                              hipStream_t stream) {
  const float* embs1     = (const float*)d_in[0];
  const float* embs2     = (const float*)d_in[1];
  const float* m_embs    = (const float*)d_in[2];
  const int*   edge      = (const int*)d_in[3];
  const float* he_init   = (const float*)d_in[4];
  const float* a_in_w    = (const float*)d_in[5];
  const float* a_in_b    = (const float*)d_in[6];
  const float* a_out_w   = (const float*)d_in[7];
  const float* a_out_b   = (const float*)d_in[8];
  const float* conv_w    = (const float*)d_in[9];
  const float* conv_b    = (const float*)d_in[10];
  const float* bn_gamma  = (const float*)d_in[11];
  const float* bn_beta   = (const float*)d_in[12];
  const float* m_in_w    = (const float*)d_in[13];
  const float* m_in_b    = (const float*)d_in[14];
  const float* m_out_w   = (const float*)d_in[15];
  const float* m_out_b   = (const float*)d_in[16];

  const int N1 = in_sizes[0] / HID_;
  const int N  = N1 + in_sizes[1] / HID_;   // 4096
  const int M  = in_sizes[2] / HID_;        // 2048
  const int E  = in_sizes[3] / 2;           // 65536
  const float SCALE = 0.17677669529663689f; // 1/sqrt(32)

  size_t off = 0;
  auto alloc = [&](size_t bytes) -> void* {
    off = (off + 255) & ~(size_t)255;
    void* p = (char*)d_ws + off;
    off += bytes;
    return p;
  };

  float* x    = (float*)alloc((size_t)N * HID_ * 4);
  float* sq   = (float*)alloc((size_t)N * 4);
  float* hf   = (float*)alloc((size_t)N * HID_ * 4);
  float* xw   = (float*)alloc((size_t)N * HID_ * 4);
  float* ef   = (float*)alloc((size_t)N * HID_ * 4);
  float* xc   = (float*)alloc((size_t)N * HID_ * 4);
  float* degD = (float*)alloc((size_t)N * 4);
  float* degB = (float*)alloc((size_t)N * 4);
  float* mean = (float*)alloc(HID_ * 4);
  float* var  = (float*)alloc(HID_ * 4);
  int*   nn   = (int*)alloc((size_t)N * KNN_ * 4);

  _Float16* xh   = (_Float16*)alloc((size_t)N * HID_ * 2);
  _Float16* heh  = (_Float16*)alloc((size_t)N * HID_ * 2);
  _Float16* mh   = (_Float16*)alloc((size_t)M * HID_ * 2);
  _Float16* w1h  = (_Float16*)alloc((size_t)3 * HID_ * HID_ * 2);
  _Float16* w1oh = (_Float16*)alloc((size_t)HID_ * HID_ * 2);
  _Float16* cwh  = (_Float16*)alloc((size_t)HID_ * HID_ * 2);
  _Float16* w2h  = (_Float16*)alloc((size_t)3 * HID_ * HID_ * 2);
  _Float16* w2oh = (_Float16*)alloc((size_t)HID_ * HID_ * 2);
  _Float16* qh   = (_Float16*)alloc((size_t)N * HID_ * 2);
  _Float16* kh   = (_Float16*)alloc((size_t)N * HID_ * 2);
  _Float16* vh   = (_Float16*)alloc((size_t)N * HID_ * 2);
  _Float16* ho   = (_Float16*)alloc((size_t)N * HID_ * 2);
  _Float16* xh2  = (_Float16*)alloc((size_t)N * HID_ * 2);
  _Float16* xh3  = (_Float16*)alloc((size_t)N * HID_ * 2);
  _Float16* qh2  = (_Float16*)alloc((size_t)M * HID_ * 2);
  _Float16* kh2  = (_Float16*)alloc((size_t)N * HID_ * 2);
  _Float16* vh2  = (_Float16*)alloc((size_t)N * HID_ * 2);
  _Float16* ho2  = (_Float16*)alloc((size_t)M * HID_ * 2);

  const int NH = N * HID_, MH = M * HID_, WH = HID_ * HID_;

  // Stage 0: concat, f16 copies, squared norms, weight conversions
  concat_kernel<<<N, HID_, 0, stream>>>(embs1, embs2, x, N1);
  cvt_f16_kernel<<<(NH + 255) / 256, 256, 0, stream>>>(x, xh, NH);
  row_sq_kernel<<<N, HID_, 0, stream>>>(x, sq);
  cvt_f16_kernel<<<(NH + 255) / 256, 256, 0, stream>>>(he_init, heh, NH);
  cvt_f16_kernel<<<(MH + 255) / 256, 256, 0, stream>>>(m_embs, mh, MH);
  cvt_f16_kernel<<<(3 * WH + 255) / 256, 256, 0, stream>>>(a_in_w, w1h, 3 * WH);
  cvt_f16_kernel<<<(WH + 255) / 256, 256, 0, stream>>>(a_out_w, w1oh, WH);
  cvt_f16_kernel<<<(WH + 255) / 256, 256, 0, stream>>>(conv_w, cwh, WH);
  cvt_f16_kernel<<<(3 * WH + 255) / 256, 256, 0, stream>>>(m_in_w, w2h, 3 * WH);
  cvt_f16_kernel<<<(WH + 255) / 256, 256, 0, stream>>>(m_out_w, w2oh, WH);

  // Stage 1: dynamic kNN (WMMA gram + streaming top-10)
  knn_topk_kernel<<<N / 16, 32, 0, stream>>>(xh, sq, nn, N);

  // Stage 2: MHA1 (hyperedge queries over node keys/values)
  dim3 gN(N / 16, 2), gM(M / 16, 2);
  gemm128_kernel<true><<<gN, 32, 0, stream>>>(heh, w1h,          a_in_b,          qh);
  gemm128_kernel<true><<<gN, 32, 0, stream>>>(xh,  w1h + WH,     a_in_b + HID_,   kh);
  gemm128_kernel<true><<<gN, 32, 0, stream>>>(xh,  w1h + 2 * WH, a_in_b + 2*HID_, vh);
  flash_attn_kernel<<<dim3(N / 16, NHEADS_), 32, 0, stream>>>(qh, kh, vh, ho, N, SCALE);
  gemm128_kernel<false><<<gN, 32, 0, stream>>>(ho, w1oh, a_out_b, hf);

  // Stage 3: x += sum over K nearest neighbours of hf
  add_knn_kernel<<<N, HID_, 0, stream>>>(x, hf, nn);

  // Stage 4: hypergraph conv  D^-1 H B^-1 H^T (X Theta) + b
  cvt_f16_kernel<<<(NH + 255) / 256, 256, 0, stream>>>(x, xh2, NH);
  gemm128_kernel<false><<<gN, 32, 0, stream>>>(xh2, cwh, nullptr, xw);
  zero_kernel<<<(NH + 255) / 256, 256, 0, stream>>>(ef, NH);
  zero_kernel<<<(NH + 255) / 256, 256, 0, stream>>>(xc, NH);
  zero_kernel<<<(N + 255) / 256, 256, 0, stream>>>(degD, N);
  zero_kernel<<<(N + 255) / 256, 256, 0, stream>>>(degB, N);
  deg_kernel<<<(E + 255) / 256, 256, 0, stream>>>(edge, degD, degB, E);
  scatter_n2e_kernel<<<E, HID_, 0, stream>>>(xw, edge, E, ef);
  scale_rows_kernel<<<N, HID_, 0, stream>>>(ef, degB);
  scatter_e2n_kernel<<<E, HID_, 0, stream>>>(ef, edge, E, xc);
  scale_bias_kernel<<<N, HID_, 0, stream>>>(xc, degD, conv_b);

  // Stage 5: BatchNorm (batch stats) + ELU -> f16
  bn_stats_kernel<<<HID_, 256, 0, stream>>>(xc, N, mean, var);
  bn_elu_kernel<<<N, HID_, 0, stream>>>(xc, xh3, mean, var, bn_gamma, bn_beta);

  // Stage 6: MHA2 (m_embs queries over node keys/values) -> d_out
  gemm128_kernel<true><<<gM, 32, 0, stream>>>(mh,  w2h,          m_in_b,          qh2);
  gemm128_kernel<true><<<gN, 32, 0, stream>>>(xh3, w2h + WH,     m_in_b + HID_,   kh2);
  gemm128_kernel<true><<<gN, 32, 0, stream>>>(xh3, w2h + 2 * WH, m_in_b + 2*HID_, vh2);
  flash_attn_kernel<<<dim3(M / 16, NHEADS_), 32, 0, stream>>>(qh2, kh2, vh2, ho2, N, SCALE);
  gemm128_kernel<false><<<gM, 32, 0, stream>>>(ho2, w2oh, m_out_b, (float*)d_out);

  (void)n_in; (void)out_size; (void)ws_size;
}
